// IDM_31456340476626
// MI455X (gfx1250) — matrix operand order repaired and checked by reading, not compiled, not run on hardware
//
#include <hip/hip_runtime.h>

// ---------------- types ----------------
typedef __attribute__((ext_vector_type(16))) __bf16 v16bf;
typedef __attribute__((ext_vector_type(8)))  float  v8f;
typedef int int4v __attribute__((vector_size(16)));

#define NH_  12
#define HD_  64
#define D_   768
#define SCALE_ 0.125f   // 1/sqrt(64)

// GEMM epilogue / input flags
#define GF_GELU     1
#define GF_RESID    2
#define GF_OUT_BF16 4
#define GF_A_BF16   8

#if defined(__gfx1250__) && __has_builtin(__builtin_amdgcn_global_load_async_to_lds_b128) && __has_builtin(__builtin_amdgcn_s_wait_asynccnt)
#define USE_ASYNC_LDS 1
#else
#define USE_ASYNC_LDS 0
#endif

__device__ __forceinline__ float gelu_f(float x) {
    float x3 = x * x * x;
    return 0.5f * x * (1.f + tanhf(0.7978845608028654f * (x + 0.044715f * x3)));
}

#if USE_ASYNC_LDS
__device__ __forceinline__ void async_copy16B(const void* g, void* l) {
    typedef __attribute__((address_space(1))) int4v* gp_t;
    typedef __attribute__((address_space(3))) int4v* lp_t;
    __builtin_amdgcn_global_load_async_to_lds_b128(
        (gp_t)(uintptr_t)g,
        (lp_t)(unsigned)(uintptr_t)l, 0, 0);
}
#endif

__device__ __forceinline__ void cvt16(float4 f0, float4 f1, float4 f2, float4 f3, __bf16* h) {
    h[0]  = (__bf16)f0.x; h[1]  = (__bf16)f0.y; h[2]  = (__bf16)f0.z; h[3]  = (__bf16)f0.w;
    h[4]  = (__bf16)f1.x; h[5]  = (__bf16)f1.y; h[6]  = (__bf16)f1.z; h[7]  = (__bf16)f1.w;
    h[8]  = (__bf16)f2.x; h[9]  = (__bf16)f2.y; h[10] = (__bf16)f2.z; h[11] = (__bf16)f2.w;
    h[12] = (__bf16)f3.x; h[13] = (__bf16)f3.y; h[14] = (__bf16)f3.z; h[15] = (__bf16)f3.w;
}

// ---------------- generic WMMA GEMM ----------------
// C[M,N] = epilogue( A[M,K] * W[K,N] + bias ),  A f32 or bf16, W f32.
// REQUIRES: K % 32 == 0, N % 64 == 0 (true for all call sites). Only M ragged.
// block = 128 threads (4 waves), tile 64(M) x 64(N), BK=32
__global__ __launch_bounds__(128)
void gemm_wmma(const float* __restrict__ Af32, const __bf16* __restrict__ Abf,
               const float* __restrict__ W, const float* __restrict__ bias,
               float* Cf, __bf16* Cb,
               const float* __restrict__ posAdd, int posMod,
               int M, int N, int K, int flags)
{
    // pads: As row = 80B (16B aligned, 20-dword stride), Bs row = 136B (8B aligned, 34-dword stride)
    __shared__ __align__(16) __bf16 As[64][40];
    __shared__ __align__(16) __bf16 Bs[32][68];

    const int bm = blockIdx.y * 64;
    const int bn = blockIdx.x * 64;
    const int tid  = threadIdx.x;
    const int wave = tid >> 5;
    const int lane = tid & 31;
    const int wm = (wave >> 1) * 32;   // wave's 32x32 subtile
    const int wn = (wave & 1)  * 32;
    const int laneM    = lane & 15;
    const int laneHalf = (lane >> 4) & 1;

    // staging assignments
    const int arow = tid >> 1;          // 0..63
    const int ac0  = (tid & 1) * 16;    // 0 / 16
    const int brow = tid >> 2;          // 0..31
    const int bc0  = (tid & 3) * 16;    // 0/16/32/48

    const int gm = bm + arow;
    const bool aok = (gm < M);
    const bool abf = (flags & GF_A_BF16) != 0;

    v8f acc[2][2] = {};

    for (int k0 = 0; k0 < K; k0 += 32) {
        __syncthreads();
        // ---- stage A tile 64x32 (row-guarded only; dead rows masked at store) ----
        if (aok) {
            if (abf) {
                const __bf16* gp = Abf + (size_t)gm * K + (k0 + ac0);
#if USE_ASYNC_LDS
                async_copy16B(gp,     &As[arow][ac0]);
                async_copy16B(gp + 8, &As[arow][ac0 + 8]);
#else
                const uint4* g4 = (const uint4*)gp;
                uint4* l4 = (uint4*)(&As[arow][ac0]);
                l4[0] = g4[0]; l4[1] = g4[1];
#endif
            } else {
                const float4* g4 = (const float4*)(Af32 + (size_t)gm * K + (k0 + ac0));
                float4 f0 = g4[0], f1 = g4[1], f2 = g4[2], f3 = g4[3];
                union { __bf16 h[16]; uint4 q[2]; } t;
                cvt16(f0, f1, f2, f3, t.h);
                uint4* l4 = (uint4*)(&As[arow][ac0]);
                l4[0] = t.q[0]; l4[1] = t.q[1];
            }
        }
        // ---- stage B tile 32x64 (unguarded: K%32==0, N%64==0) ----
        {
            const float4* g4 = (const float4*)(W + (size_t)(k0 + brow) * N + (bn + bc0));
            float4 f0 = g4[0], f1 = g4[1], f2 = g4[2], f3 = g4[3];
            union { __bf16 h[16]; uint2 q[4]; } t;
            cvt16(f0, f1, f2, f3, t.h);
            uint2* l2 = (uint2*)(&Bs[brow][bc0]);
            l2[0] = t.q[0]; l2[1] = t.q[1]; l2[2] = t.q[2]; l2[3] = t.q[3];
        }
        // prefetch next W K-tile (global_prefetch)
        if (k0 + 32 < K)
            __builtin_prefetch(W + (size_t)(k0 + 32 + brow) * N + (bn + bc0), 0, 0);
#if USE_ASYNC_LDS
        if (abf && aok) __builtin_amdgcn_s_wait_asynccnt(0);
#endif
        __syncthreads();

        // ---- fragments (ISA 7.12.2 layouts) ----
        v16bf afrag[2], bfrag[2];
        #pragma unroll
        for (int t = 0; t < 2; ++t) {
            int mrow = wm + t * 16 + laneM;
            #pragma unroll
            for (int j = 0; j < 16; ++j) {
                int kk = (j >> 3) * 16 + laneHalf * 8 + (j & 7);
                afrag[t][j] = As[mrow][kk];
            }
            #pragma unroll
            for (int j = 0; j < 16; ++j)
                bfrag[t][j] = Bs[(j >> 3) * 16 + laneHalf * 8 + (j & 7)][wn + t * 16 + laneM];
        }
        #pragma unroll
        for (int tm = 0; tm < 2; ++tm)
            #pragma unroll
            for (int tn = 0; tn < 2; ++tn)
                acc[tm][tn] = __builtin_amdgcn_wmma_f32_16x16x32_bf16(
                    false, afrag[tm], false, bfrag[tn],
                    (short)0, acc[tm][tn], false, false);
    }

    // epilogue: VGPR r -> M = r + 8*(lane>>4); N = lane&15
    #pragma unroll
    for (int tm = 0; tm < 2; ++tm)
        #pragma unroll
        for (int tn = 0; tn < 2; ++tn)
            #pragma unroll
            for (int r = 0; r < 8; ++r) {
                int m = bm + wm + tm * 16 + r + 8 * laneHalf;
                int n = bn + wn + tn * 16 + laneM;
                if (m < M && n < N) {
                    float v = acc[tm][tn][r];
                    if (bias)   v += bias[n];
                    if (posAdd) v += posAdd[(size_t)(m % posMod) * N + n];
                    if (flags & GF_GELU)  v = gelu_f(v);
                    if (flags & GF_RESID) v += Cf[(size_t)m * N + n];
                    if (flags & GF_OUT_BF16) Cb[(size_t)m * N + n] = (__bf16)v;
                    else                     Cf[(size_t)m * N + n] = v;
                }
            }
}

// ---------------- sincos 3D positional embedding ----------------
__global__ __launch_bounds__(256)
void pos_embed_kernel(float* out, int rows, int tstart)
{
    int i = blockIdx.x * 256 + threadIdx.x;
    if (i >= rows * D_) return;
    int row = i / D_, col = i % D_;
    int t  = tstart + row / 256;   // H*W = 256
    int hh = (row / 16) & 15;
    int ww = row & 15;
    int axis   = col >> 8;
    int within = col & 255;
    int half   = within >> 7;
    int fi     = within & 127;
    float omega = __expf(-0.07195578042f * (float)fi);  // 10000^(-fi/128)
    float p = (axis == 0) ? (float)t : (axis == 1) ? (float)hh : (float)ww;
    float ang = p * omega;
    out[i] = half ? __cosf(ang) : __sinf(ang);
}

// ---------------- q init: broadcast queries + pos emb ----------------
__global__ __launch_bounds__(256)
void qinit_kernel(const float* aq, const float* ape, float* q, int total, int per)
{
    int i = blockIdx.x * 256 + threadIdx.x;
    if (i < total) { int j = i % per; q[i] = aq[j] + ape[j]; }
}

// ---------------- LayerNorm (row of 768) -> bf16 ----------------
__global__ __launch_bounds__(256)
void ln_to_bf16(const float* __restrict__ x, const float* __restrict__ g,
                const float* __restrict__ b, __bf16* y, int N)
{
    __shared__ float red[256];
    int row = blockIdx.x, tid = threadIdx.x;
    const float* xr = x + (size_t)row * N;
    float s = 0.f;
    for (int i = tid; i < N; i += 256) s += xr[i];
    red[tid] = s; __syncthreads();
    for (int off = 128; off > 0; off >>= 1) {
        if (tid < off) red[tid] += red[tid + off];
        __syncthreads();
    }
    float mu = red[0] / N;
    __syncthreads();
    float v = 0.f;
    for (int i = tid; i < N; i += 256) { float d = xr[i] - mu; v += d * d; }
    red[tid] = v; __syncthreads();
    for (int off = 128; off > 0; off >>= 1) {
        if (tid < off) red[tid] += red[tid + off];
        __syncthreads();
    }
    float rstd = rsqrtf(red[0] / N + 1e-6f);
    __bf16* yr = y + (size_t)row * N;
    for (int i = tid; i < N; i += 256)
        yr[i] = (__bf16)((xr[i] - mu) * rstd * g[i] + b[i]);
}

// ---------------- attention: one block per (b, head, query) ----------------
__global__ __launch_bounds__(256)
void attn_kernel(const float* __restrict__ qh, const float* __restrict__ kh,
                 const float* __restrict__ vh, __bf16* o, int Lq, int Lk)
{
    __shared__ float sc[2048];
    __shared__ float red[256];
    __shared__ float s_max, s_sum;
    int idx = blockIdx.x;
    int qi = idx % Lq; idx /= Lq;
    int h  = idx % NH_; idx /= NH_;
    int b  = idx;
    int tid = threadIdx.x;
    const float* qrow = qh + (size_t)(b * Lq + qi) * D_ + h * HD_;

    float lmax = -1e30f;
    for (int k = tid; k < Lk; k += 256) {
        const float* krow = kh + (size_t)(b * Lk + k) * D_ + h * HD_;
        float d = 0.f;
        #pragma unroll 8
        for (int j = 0; j < HD_; ++j) d += qrow[j] * krow[j];
        d *= SCALE_;
        sc[k] = d;
        lmax = fmaxf(lmax, d);
    }
    red[tid] = lmax; __syncthreads();
    for (int off = 128; off > 0; off >>= 1) {
        if (tid < off) red[tid] = fmaxf(red[tid], red[tid + off]);
        __syncthreads();
    }
    if (tid == 0) s_max = red[0];
    __syncthreads();
    float m = s_max, lsum = 0.f;
    for (int k = tid; k < Lk; k += 256) { float e = __expf(sc[k] - m); sc[k] = e; lsum += e; }
    red[tid] = lsum; __syncthreads();
    for (int off = 128; off > 0; off >>= 1) {
        if (tid < off) red[tid] += red[tid + off];
        __syncthreads();
    }
    if (tid == 0) s_sum = red[0];
    __syncthreads();
    float inv = 1.f / s_sum;
    int d = tid & 63, chunk = tid >> 6;
    float acc = 0.f;
    for (int k = chunk; k < Lk; k += 4)
        acc += sc[k] * vh[(size_t)(b * Lk + k) * D_ + h * HD_ + d];
    red[tid] = acc; __syncthreads();
    if (chunk == 0) {
        float t = red[d] + red[d + 64] + red[d + 128] + red[d + 192];
        o[(size_t)(b * Lq + qi) * D_ + h * HD_ + d] = (__bf16)(t * inv);
    }
}

// ---------------- host orchestration ----------------
extern "C" void kernel_launch(void* const* d_in, const int* in_sizes, int n_in,
                              void* d_out, int out_size, void* d_ws, size_t ws_size,
                              hipStream_t stream)
{
    const float* past_frames   = (const float*)d_in[0];
    const float* future_frames = (const float*)d_in[1];
    const float* state_in      = (const float*)d_in[2];
    const float* aux_frames    = (const float*)d_in[3];

    // Param ordering: insertion order vs pytree-alphabetical order.
    bool ins = (in_sizes[4] == 768 * 768);   // past_W first (insertion) vs action_pos_emb (6144)
    auto g = [&](int insIdx, int alphaIdx) -> const float* {
        return (const float*)d_in[4 + (ins ? insIdx : alphaIdx)];
    };
    const float* past_W  = g(0, 28);  const float* past_b  = g(1, 29);
    const float* fut_W   = g(2, 16);  const float* fut_b   = g(3, 17);
    const float* state_W = g(4, 40);  const float* state_b = g(5, 41);
    const float* aux_W1  = g(6, 2);   const float* aux_b1  = g(7, 4);
    const float* aux_W2  = g(8, 3);   const float* aux_b2  = g(9, 5);
    const float* action_queries = g(10, 1);
    const float* action_pos_emb = g(11, 0);
    const float* ca_ln_g = g(12, 15); const float* ca_ln_b = g(13, 14);
    const float* sa_ln_g = g(14, 39); const float* sa_ln_b = g(15, 38);
    const float* mlp_ln_g = g(16, 23); const float* mlp_ln_b = g(17, 22);
    const float* mlp_W1 = g(18, 18);  const float* mlp_b1 = g(19, 20);
    const float* mlp_W2 = g(20, 19);  const float* mlp_b2 = g(21, 21);
    const float* out_ln_g = g(22, 27); const float* out_ln_b = g(23, 26);
    const float* out_W = g(24, 24);   const float* out_b = g(25, 25);
    const float* ca_Wq = g(26, 8);  const float* sa_Wq = g(27, 32);
    const float* ca_Wk = g(28, 6);  const float* sa_Wk = g(29, 30);
    const float* ca_Wv = g(30, 9);  const float* sa_Wv = g(31, 33);
    const float* ca_Wo = g(32, 7);  const float* sa_Wo = g(33, 31);
    const float* ca_bq = g(34, 12); const float* sa_bq = g(35, 36);
    const float* ca_bk = g(36, 10); const float* sa_bk = g(37, 34);
    const float* ca_bv = g(38, 13); const float* sa_bv = g(39, 37);
    const float* ca_bo = g(40, 11); const float* sa_bo = g(41, 35);

    // workspace bump allocator
    char* wp = (char*)d_ws;
    auto alloc = [&](size_t bytes) -> void* {
        void* p = wp; wp += (bytes + 255) & ~(size_t)255; return p;
    };
    float*  pastPos  = (float*) alloc(1024 * 768 * 4);
    float*  futPos   = (float*) alloc(2048 * 768 * 4);
    __bf16* srcPast  = (__bf16*)alloc((size_t)4096 * 768 * 2);
    __bf16* srcFut   = (__bf16*)alloc((size_t)8192 * 768 * 2);
    __bf16* srcState = (__bf16*)alloc(4 * 768 * 2);
    __bf16* srcAux   = (__bf16*)alloc((size_t)8192 * 768 * 2);
    __bf16* tmpAux   = (__bf16*)alloc((size_t)8192 * 768 * 2);
    float*  qbuf     = (float*) alloc(32 * 768 * 4);
    __bf16* qn       = (__bf16*)alloc(32 * 768 * 2);
    float*  qhb      = (float*) alloc(32 * 768 * 4);
    float*  khb      = (float*) alloc((size_t)8192 * 768 * 4);
    float*  vhb      = (float*) alloc((size_t)8192 * 768 * 4);
    __bf16* ob       = (__bf16*)alloc(32 * 768 * 2);
    __bf16* hb       = (__bf16*)alloc(32 * 3072 * 2);

    auto gemm = [&](const void* A, const float* W_, const float* bias_,
                    float* Cf, __bf16* Cb, const float* pos, int posMod,
                    int M, int N, int K, int flags) {
        dim3 grid((N + 63) / 64, (M + 63) / 64);
        gemm_wmma<<<grid, 128, 0, stream>>>(
            (flags & GF_A_BF16) ? nullptr : (const float*)A,
            (flags & GF_A_BF16) ? (const __bf16*)A : nullptr,
            W_, bias_, Cf, Cb, pos, posMod > 0 ? posMod : 1, M, N, K, flags);
    };

    // positional embeddings
    pos_embed_kernel<<<(1024 * 768 + 255) / 256, 256, 0, stream>>>(pastPos, 1024, 0);
    pos_embed_kernel<<<(2048 * 768 + 255) / 256, 256, 0, stream>>>(futPos, 2048, 4);

    // source token prep (stored bf16, reused for K/V in all 6 layers)
    gemm(past_frames,   past_W,  past_b,  nullptr, srcPast,  pastPos, 1024, 4096, 768, 768, GF_OUT_BF16);
    gemm(future_frames, fut_W,   fut_b,   nullptr, srcFut,   futPos,  2048, 8192, 768, 768, GF_OUT_BF16);
    gemm(state_in,      state_W, state_b, nullptr, srcState, nullptr, 0,    4,    768, 64,  GF_OUT_BF16);
    gemm(aux_frames,    aux_W1,  aux_b1,  nullptr, tmpAux,   nullptr, 0,    8192, 768, 768, GF_OUT_BF16 | GF_GELU);
    gemm(tmpAux,        aux_W2,  aux_b2,  nullptr, srcAux,   pastPos, 1024, 8192, 768, 768, GF_OUT_BF16 | GF_A_BF16);

    // q = action_queries + action_pos_emb, broadcast over batch
    qinit_kernel<<<(32 * 768 + 255) / 256, 256, 0, stream>>>(
        action_queries, action_pos_emb, qbuf, 32 * 768, 8 * 768);

    struct Src { const __bf16* p; int Lk; };
    Src srcs[4] = { {srcPast, 1024}, {srcFut, 2048}, {srcState, 1}, {srcAux, 2048} };

    const size_t WSZ = (size_t)768 * 768;
    for (int d = 0; d < 6; ++d) {
        for (int s = 0; s < 4; ++s) {
            size_t ds = (size_t)(d * 4 + s);
            ln_to_bf16<<<32, 256, 0, stream>>>(qbuf, ca_ln_g + ds * 768, ca_ln_b + ds * 768, qn, 768);
            gemm(qn, ca_Wq + ds * WSZ, ca_bq + ds * 768, qhb, nullptr, nullptr, 0, 32, 768, 768, GF_A_BF16);
            int Lk = srcs[s].Lk;
            gemm(srcs[s].p, ca_Wk + ds * WSZ, ca_bk + ds * 768, khb, nullptr, nullptr, 0, 4 * Lk, 768, 768, GF_A_BF16);
            gemm(srcs[s].p, ca_Wv + ds * WSZ, ca_bv + ds * 768, vhb, nullptr, nullptr, 0, 4 * Lk, 768, 768, GF_A_BF16);
            attn_kernel<<<4 * NH_ * 8, 256, 0, stream>>>(qhb, khb, vhb, ob, 8, Lk);
            gemm(ob, ca_Wo + ds * WSZ, ca_bo + ds * 768, qbuf, nullptr, nullptr, 0, 32, 768, 768, GF_A_BF16 | GF_RESID);
        }
        // self-attention (kv = qn, Lk = 8)
        ln_to_bf16<<<32, 256, 0, stream>>>(qbuf, sa_ln_g + d * 768, sa_ln_b + d * 768, qn, 768);
        gemm(qn, sa_Wq + d * WSZ, sa_bq + d * 768, qhb, nullptr, nullptr, 0, 32, 768, 768, GF_A_BF16);
        gemm(qn, sa_Wk + d * WSZ, sa_bk + d * 768, khb, nullptr, nullptr, 0, 32, 768, 768, GF_A_BF16);
        gemm(qn, sa_Wv + d * WSZ, sa_bv + d * 768, vhb, nullptr, nullptr, 0, 32, 768, 768, GF_A_BF16);
        attn_kernel<<<4 * NH_ * 8, 256, 0, stream>>>(qhb, khb, vhb, ob, 8, 8);
        gemm(ob, sa_Wo + d * WSZ, sa_bo + d * 768, qbuf, nullptr, nullptr, 0, 32, 768, 768, GF_A_BF16 | GF_RESID);
        // MLP
        ln_to_bf16<<<32, 256, 0, stream>>>(qbuf, mlp_ln_g + d * 768, mlp_ln_b + d * 768, qn, 768);
        gemm(qn, mlp_W1 + (size_t)d * 768 * 3072, mlp_b1 + d * 3072, nullptr, hb, nullptr, 0,
             32, 3072, 768, GF_A_BF16 | GF_OUT_BF16 | GF_GELU);
        gemm(hb, mlp_W2 + (size_t)d * 3072 * 768, mlp_b2 + d * 768, qbuf, nullptr, nullptr, 0,
             32, 768, 3072, GF_A_BF16 | GF_RESID);
    }
    // output head
    ln_to_bf16<<<32, 256, 0, stream>>>(qbuf, out_ln_g, out_ln_b, qn, 768);
    gemm(qn, out_W, out_b, (float*)d_out, nullptr, nullptr, 0, 32, 64, 768, GF_A_BF16);
}
